// GATModel_67001489818088
// MI455X (gfx1250) — compile-verified, matrix operand order checked by reading, not run on hardware
//
#include <hip/hip_runtime.h>
#include <hip/hip_bf16.h>
#include <math.h>

typedef __attribute__((ext_vector_type(16))) __bf16 v16bf;
typedef __attribute__((ext_vector_type(8)))  float  v8f;
typedef __attribute__((ext_vector_type(4)))  unsigned int v4u;
typedef __attribute__((ext_vector_type(8)))  int v8i;
typedef __attribute__((ext_vector_type(4)))  int v4i;
typedef unsigned short u16;
typedef unsigned int   u32;
typedef unsigned long long u64;

#define DIVUP(a,b) (((a)+(b)-1)/(b))

#if defined(__has_builtin)
#  if __has_builtin(__builtin_amdgcn_tensor_load_to_lds) && __has_builtin(__builtin_amdgcn_s_wait_tensorcnt)
#    define HAS_TDM 1
#  else
#    define HAS_TDM 0
#  endif
#else
#  define HAS_TDM 0
#endif

// ---------- small helpers ----------
__device__ __forceinline__ u16 f2bf(float f) {          // round-to-nearest-even f32->bf16
  u32 u = __float_as_uint(f);
  u32 r = u + 0x7fffu + ((u >> 16) & 1u);
  return (u16)(r >> 16);
}
__device__ __forceinline__ u32 f2ord(float f) {         // order-preserving uint encoding
  u32 u = __float_as_uint(f);
  return (u & 0x80000000u) ? ~u : (u | 0x80000000u);
}
__device__ __forceinline__ float ord2f(u32 u) {
  u32 v = (u & 0x80000000u) ? (u & 0x7fffffffu) : ~u;
  return __uint_as_float(v);
}

// ---------- WMMA GEMM: C[M,Nact](f32,ldc) = A[M,Kp](bf16) @ B[Kp,Npad](bf16) ----------
// Block tile 256x128, 8 waves (4x2), each wave 64x64 = 16 WMMA accumulators.
// A tile staged by the Tensor Data Mover (tensor_load_to_lds) when available.
struct Bf16x16 { uint4 lo, hi; };
__device__ __forceinline__ v16bf frag_from_lds(const u16* p) {
  Bf16x16 t;
  t.lo = *(const uint4*)(p);
  t.hi = *(const uint4*)(p + 16);
  return __builtin_bit_cast(v16bf, t);
}

__global__ __launch_bounds__(256)
void k_gemm_bf16(const u16* __restrict__ A, const u16* __restrict__ B,
                 float* __restrict__ C, int M, int Nact, int Npad, int Kp, int ldc)
{
  __shared__ __align__(16) u16 ldsA[256 * 32];   // 256 rows x 32 k (16 KB)
  __shared__ __align__(16) u16 ldsBt[128 * 32];  // transposed [n][k] (8 KB)
  const int tid  = threadIdx.x;
  const int lane = tid & 31, wid = tid >> 5;
  const int wm = wid >> 1, wn = wid & 1;         // 4x2 wave grid, each wave 64x64
  const int mbase = blockIdx.y * 256, nbase = blockIdx.x * 128;
  const int l15 = lane & 15, lh = (lane >> 4) * 8;

  v8f acc[4][4];
  #pragma unroll
  for (int mi = 0; mi < 4; ++mi)
    #pragma unroll
    for (int ni = 0; ni < 4; ++ni) acc[mi][ni] = {};

  for (int k0 = 0; k0 < Kp; k0 += 32) {
    __syncthreads();   // previous compute done; LDS safe to overwrite

#if HAS_TDM
    if (wid == 0) {
      // TDM descriptor: 2D tile (32 k) x (256 rows), 2-byte elements, linear LDS dest.
      unsigned lds_base = (unsigned)(uintptr_t)(&ldsA[0]);          // low 32b = LDS offset
      u64 ga = (u64)(uintptr_t)A + ((u64)(unsigned)mbase * (unsigned)Kp + (unsigned)k0) * 2ull;
      unsigned rows_rem = (unsigned)(M - mbase);                    // OOB rows read as zero
      v4u g0;
      g0.x = 1u;                                                    // count=1, no gather
      g0.y = lds_base;                                              // lds_addr
      g0.z = (unsigned)ga;                                          // global_addr lo
      g0.w = (unsigned)((ga >> 32) & 0x1FFFFFFu) | (2u << 30);      // global_addr hi | type=2
      v8i g1;
      g1[0] = (int)(1u << 16);                                      // data_size=1 (2 bytes)
      g1[1] = (int)(((unsigned)Kp & 0xFFFFu) << 16);                // tensor_dim0 lo16
      g1[2] = (int)((((unsigned)Kp >> 16) & 0xFFFFu) | ((rows_rem & 0xFFFFu) << 16)); // dim0 hi | dim1 lo
      g1[3] = (int)(((rows_rem >> 16) & 0xFFFFu) | (32u << 16));    // dim1 hi | tile_dim0=32
      g1[4] = 256;                                                  // tile_dim1=256 (tile_dim2=0)
      g1[5] = Kp;                                                   // tensor_dim0_stride lo32
      g1[6] = 0;
      g1[7] = 0;
      v4i z4 = {0, 0, 0, 0};
#if __clang_major__ >= 23
      v8i z8 = {0, 0, 0, 0, 0, 0, 0, 0};
      __builtin_amdgcn_tensor_load_to_lds(g0, g1, z4, z4, z8, 0);
#else
      __builtin_amdgcn_tensor_load_to_lds(g0, g1, z4, z4, 0);
#endif
    }
#else
    // manual A staging: 256x32 bf16 = 4 x (256 threads x 16B)
    #pragma unroll
    for (int it = 0; it < 4; ++it) {
      int idx = it * 256 + tid;
      int row = idx >> 2, ch = (idx & 3) * 8;
      uint4 v; v.x = v.y = v.z = v.w = 0u;
      int gm = mbase + row;
      if (gm < M) v = *(const uint4*)(A + (size_t)gm * Kp + k0 + ch);
      *(uint4*)(ldsA + row * 32 + ch) = v;
    }
#endif

    // stage B tile transposed (32k x 128n -> [n][k]); 2 x (256 threads x 8 elems)
    #pragma unroll
    for (int it = 0; it < 2; ++it) {
      int idx = it * 256 + tid;
      int kk = idx >> 4, n0 = (idx & 15) * 8;
      uint4 v = *(const uint4*)(B + (size_t)(k0 + kk) * Npad + nbase + n0);
      const u16* pv = (const u16*)&v;
      #pragma unroll
      for (int i = 0; i < 8; ++i) ldsBt[(n0 + i) * 32 + kk] = pv[i];
    }
    if (k0 + 32 < Kp) {   // hint next B tile -> global_prefetch_b8
      int kk = tid >> 4, n0 = (tid & 15) * 8;
      __builtin_prefetch(B + (size_t)(k0 + 32 + kk) * Npad + nbase + n0, 0, 1);
    }

#if HAS_TDM
    if (wid == 0) __builtin_amdgcn_s_wait_tensorcnt(0);
#endif
    __syncthreads();

    v16bf af[4], bfr[4];
    #pragma unroll
    for (int mi = 0; mi < 4; ++mi)
      af[mi] = frag_from_lds(ldsA + (wm * 64 + mi * 16 + l15) * 32 + lh);
    #pragma unroll
    for (int ni = 0; ni < 4; ++ni)
      bfr[ni] = frag_from_lds(ldsBt + (wn * 64 + ni * 16 + l15) * 32 + lh);
    #pragma unroll
    for (int mi = 0; mi < 4; ++mi)
      #pragma unroll
      for (int ni = 0; ni < 4; ++ni)
        acc[mi][ni] = __builtin_amdgcn_wmma_f32_16x16x32_bf16(
            false, af[mi], false, bfr[ni], (short)0, acc[mi][ni], false, false);
  }

  #pragma unroll
  for (int mi = 0; mi < 4; ++mi) {
    #pragma unroll
    for (int ni = 0; ni < 4; ++ni) {
      int n = nbase + wn * 64 + ni * 16 + l15;
      #pragma unroll
      for (int r = 0; r < 8; ++r) {
        int m = mbase + wm * 64 + mi * 16 + r + (lane >> 4) * 8;
        if (m < M && n < Nact) C[(size_t)m * ldc + n] = acc[mi][ni][r];
      }
    }
  }
}

// ---------- utility kernels ----------
__global__ void k_convert_pad(const float* __restrict__ src, u16* __restrict__ dst,
                              int srcR, int srcC, int dstC, long total)
{
  long t = (long)blockIdx.x * blockDim.x + threadIdx.x;
  if (t >= total) return;
  int r = (int)(t / dstC), c = (int)(t % dstC);
  u16 v = 0;
  if (r < srcR && c < srcC) v = f2bf(src[(size_t)r * srcC + c]);
  dst[t] = v;
}
__global__ void k_fill_f32(float* p, long n, float v) {
  long t = (long)blockIdx.x * blockDim.x + threadIdx.x;
  if (t < n) p[t] = v;
}
__global__ void k_fill_u32(u32* p, long n, u32 v) {
  long t = (long)blockIdx.x * blockDim.x + threadIdx.x;
  if (t < n) p[t] = v;
}
__global__ void k_counts(const int* __restrict__ batch, float* __restrict__ cnt, int n) {
  int t = blockIdx.x * blockDim.x + threadIdx.x;
  if (t < n) atomicAdd(&cnt[batch[t]], 1.0f);
}
__global__ void k_seg_add(const float* __restrict__ x, const int* __restrict__ batch,
                          float* __restrict__ acc, long n, int Cdim) {
  long t = (long)blockIdx.x * blockDim.x + threadIdx.x;
  if (t >= n) return;
  int node = (int)(t / Cdim), i = (int)(t % Cdim);
  atomicAdd(&acc[(size_t)batch[node] * Cdim + i], x[t]);
}
__global__ void k_div_cnt(float* __restrict__ acc, const float* __restrict__ cnt,
                          long n, int Cdim) {
  long t = (long)blockIdx.x * blockDim.x + threadIdx.x;
  if (t >= n) return;
  int g = (int)(t / Cdim);
  acc[t] /= fmaxf(cnt[g], 1.0f);
}
__global__ void k_bias_relu(float* __restrict__ x, const float* __restrict__ b,
                            long n, int Cdim) {
  long t = (long)blockIdx.x * blockDim.x + threadIdx.x;
  if (t < n) x[t] = fmaxf(x[t] + b[t % Cdim], 0.0f);
}
__global__ void k_bias(float* __restrict__ x, const float* __restrict__ b,
                       long n, int Cdim) {
  long t = (long)blockIdx.x * blockDim.x + threadIdx.x;
  if (t < n) x[t] += b[t % Cdim];
}

// ---------- GATv2 edge kernels ----------
__global__ __launch_bounds__(256)
void k_edge_score(const float* __restrict__ xl, const float* __restrict__ xr,
                  const float* __restrict__ ea, const float* __restrict__ We,
                  const float* __restrict__ att, const int* __restrict__ ei,
                  float* __restrict__ score, u32* __restrict__ smax,
                  int E, int H, int C)
{
  int gw = blockIdx.x * 8 + (threadIdx.x >> 5);  // one wave per (edge, head)
  if (gw >= E * H) return;
  int lane = threadIdx.x & 31;
  int e = gw / H, h = gw - e * H;
  int s = ei[e], d = ei[E + e];
  int HC = H * C;
  float eav[9];
  #pragma unroll
  for (int j = 0; j < 9; ++j) eav[j] = ea[(size_t)e * 9 + j];
  const float* pl = xl + (size_t)s * HC + h * C;
  const float* pr = xr + (size_t)d * HC + h * C;
  const float* pa = att + h * C;
  float part = 0.0f;
  for (int c = lane; c < C; c += 32) {
    float v = pl[c] + pr[c];
    #pragma unroll
    for (int j = 0; j < 9; ++j) v += eav[j] * We[(size_t)j * HC + h * C + c];
    v = v > 0.0f ? v : 0.2f * v;       // leaky relu
    part += pa[c] * v;
  }
  #pragma unroll
  for (int o = 16; o > 0; o >>= 1) part += __shfl_xor(part, o, 32);
  if (lane == 0) {
    score[(size_t)e * H + h] = part;
    atomicMax(&smax[(size_t)d * H + h], f2ord(part));
  }
}
__global__ void k_edge_expo(const float* __restrict__ score, const u32* __restrict__ smax,
                            float* __restrict__ den, float* __restrict__ expo,
                            const int* __restrict__ ei, int E, int H)
{
  long t = (long)blockIdx.x * blockDim.x + threadIdx.x;
  if (t >= (long)E * H) return;
  int e = (int)(t / H), h = (int)(t - (long)e * H);
  int d = ei[E + e];
  float ex = __expf(score[t] - ord2f(smax[(size_t)d * H + h]));
  expo[t] = ex;
  atomicAdd(&den[(size_t)d * H + h], ex);
}
__global__ __launch_bounds__(256)
void k_edge_agg(const float* __restrict__ xl, const float* __restrict__ expo,
                const float* __restrict__ den, const int* __restrict__ ei,
                float* __restrict__ out, int E, int H, int HC, int shiftC)
{
  int e = blockIdx.x;
  int s = ei[e], d = ei[E + e];
  for (int i = threadIdx.x; i < HC; i += 256) {
    int h = i >> shiftC;
    float alpha = expo[(size_t)e * H + h] / (den[(size_t)d * H + h] + 1e-16f);
    atomicAdd(&out[(size_t)d * HC + i], alpha * xl[(size_t)s * HC + i]);
  }
}

// ---------- GraphNorm ----------
__global__ void k_gn_var(const float* __restrict__ x, const int* __restrict__ batch,
                         const float* __restrict__ mean, const float* __restrict__ a,
                         float* __restrict__ gvar, long n, int Cdim)
{
  long t = (long)blockIdx.x * blockDim.x + threadIdx.x;
  if (t >= n) return;
  int node = (int)(t / Cdim), i = (int)(t % Cdim);
  int g = batch[node];
  float sub = x[t] - a[i] * mean[(size_t)g * Cdim + i];
  atomicAdd(&gvar[(size_t)g * Cdim + i], sub * sub);
}
__global__ void k_gn_apply(float* __restrict__ x, const int* __restrict__ batch,
                           const float* __restrict__ mean, const float* __restrict__ a,
                           const float* __restrict__ w, const float* __restrict__ b,
                           const float* __restrict__ gvar, const float* __restrict__ cnt,
                           long n, int Cdim)
{
  long t = (long)blockIdx.x * blockDim.x + threadIdx.x;
  if (t >= n) return;
  int node = (int)(t / Cdim), i = (int)(t % Cdim);
  int g = batch[node];
  float c = fmaxf(cnt[g], 1.0f);
  float var = gvar[(size_t)g * Cdim + i] / c;
  float sub = x[t] - a[i] * mean[(size_t)g * Cdim + i];
  x[t] = w[i] * sub * rsqrtf(var + 1e-5f) + b[i];
}

// ---------- head: BatchNorm1d (training stats) + ReLU, emit bf16 ----------
__global__ __launch_bounds__(256)
void k_bn_relu_bf16(const float* __restrict__ h, const float* __restrict__ g,
                    const float* __restrict__ b, u16* __restrict__ hbf,
                    int rows, int cols)
{
  __shared__ float red[256];
  int j = blockIdx.x, tid = threadIdx.x;
  float s = 0.0f;
  for (int r = tid; r < rows; r += 256) s += h[(size_t)r * cols + j];
  red[tid] = s; __syncthreads();
  for (int o = 128; o > 0; o >>= 1) { if (tid < o) red[tid] += red[tid + o]; __syncthreads(); }
  float mu = red[0] / rows;
  __syncthreads();
  s = 0.0f;
  for (int r = tid; r < rows; r += 256) { float d = h[(size_t)r * cols + j] - mu; s += d * d; }
  red[tid] = s; __syncthreads();
  for (int o = 128; o > 0; o >>= 1) { if (tid < o) red[tid] += red[tid + o]; __syncthreads(); }
  float inv = rsqrtf(red[0] / rows + 1e-5f);
  for (int r = tid; r < rows; r += 256) {
    float v = g[j] * (h[(size_t)r * cols + j] - mu) * inv + b[j];
    hbf[(size_t)r * cols + j] = f2bf(fmaxf(v, 0.0f));
  }
}
__global__ void k_final(float* __restrict__ out, const float* __restrict__ b,
                        long total, int cols)
{
  long t = (long)blockIdx.x * blockDim.x + threadIdx.x;
  if (t >= total) return;
  float v = out[t] + b[t % cols];
  out[t] = v;                                  // logits
  out[total + t] = 1.0f / (1.0f + __expf(-v)); // sigmoid
}

// ---------- host orchestration ----------
extern "C" void kernel_launch(void* const* d_in, const int* in_sizes, int n_in,
                              void* d_out, int out_size, void* d_ws, size_t ws_size,
                              hipStream_t stream)
{
  (void)in_sizes; (void)n_in; (void)out_size; (void)ws_size;
  const int NS = 24000, ES = 96000, FS = 69, G = 1024;

  const float* x_s  = (const float*)d_in[0];
  const int*   ei_s = (const int*)  d_in[1];
  const float* ea_s = (const float*)d_in[2];
  const int*   bt_s = (const int*)  d_in[3];
  // d_in[4..7] = t-branch inputs: dead code (output depends only on xs)
  auto F = [&](int i) { return (const float*)d_in[i]; };
  // params, insertion order: s1=8..12, s2=13..17, s3=18..22, gn1=23..25, gn2=26..28, gn3=29..31
  // t-params 32..55 unused; lin1_W=56, lin1_b=57, bn_g=58, bn_b=59, lin2_W=60, lin2_b=61

  char* wp = (char*)d_ws;
  auto alloc = [&](size_t bytes) { char* p = wp; wp += (bytes + 255) & ~(size_t)255; return p; };
  u16*   ABF   = (u16*)  alloc((size_t)NS * 1024 * 2); // bf16 activations (padded K)
  float* XL    = (float*)alloc((size_t)NS * 1024 * 4);
  float* XR    = (float*)alloc((size_t)NS * 1024 * 4); // also reused as layer output
  u16*   WBF   = (u16*)  alloc((size_t)1024 * 1408 * 2);
  float* SCORE = (float*)alloc((size_t)ES * 8 * 4);
  float* EXPO  = (float*)alloc((size_t)ES * 8 * 4);
  u32*   SMAX  = (u32*)  alloc((size_t)NS * 8 * 4);
  float* DEN   = (float*)alloc((size_t)NS * 8 * 4);
  float* GSUM  = (float*)alloc((size_t)G * 1024 * 4);
  float* GVAR  = (float*)alloc((size_t)G * 1024 * 4);
  float* CNT   = (float*)alloc((size_t)G * 4);
  float* POOL  = (float*)alloc((size_t)G * 512 * 4);
  float* FCH   = (float*)alloc((size_t)G * 1024 * 4);
  u16*   FCHBF = (u16*)  alloc((size_t)G * 1024 * 2);

  auto EW = [](long n) { return dim3((unsigned)DIVUP(n, 256L)); };
  auto GG = [](int M, int Npad) { return dim3((unsigned)(Npad / 128), (unsigned)DIVUP(M, 256)); };

  auto run_gat_layer = [&](int K, int Kp, int H, int C, int shiftC,
                           const float* Wl, const float* Wr, const float* We,
                           const float* att, const float* bias,
                           const float* gnw, const float* gnb, const float* gna)
  {
    const int HC = H * C;
    const long nW = (long)Kp * HC;
    // xl = A @ Wl ; xr = A @ Wr   (WMMA bf16)
    k_convert_pad<<<EW(nW), 256, 0, stream>>>(Wl, WBF, K, HC, HC, nW);
    k_gemm_bf16<<<GG(NS, HC), 256, 0, stream>>>(ABF, WBF, XL, NS, HC, HC, Kp, HC);
    k_convert_pad<<<EW(nW), 256, 0, stream>>>(Wr, WBF, K, HC, HC, nW);
    k_gemm_bf16<<<GG(NS, HC), 256, 0, stream>>>(ABF, WBF, XR, NS, HC, HC, Kp, HC);
    // segment softmax over incoming edges
    k_fill_u32<<<EW((long)NS * H), 256, 0, stream>>>(SMAX, (long)NS * H, 0x007FFFFFu); // ord(-inf)
    k_fill_f32<<<EW((long)NS * H), 256, 0, stream>>>(DEN, (long)NS * H, 0.0f);
    k_edge_score<<<DIVUP(ES * H, 8), 256, 0, stream>>>(XL, XR, ea_s, We, att, ei_s, SCORE, SMAX, ES, H, C);
    k_edge_expo<<<EW((long)ES * H), 256, 0, stream>>>(SCORE, SMAX, DEN, EXPO, ei_s, ES, H);
    // aggregate: OUT aliases XR (xr no longer needed)
    float* OUT = XR;
    k_fill_f32<<<EW((long)NS * HC), 256, 0, stream>>>(OUT, (long)NS * HC, 0.0f);
    k_edge_agg<<<ES, 256, 0, stream>>>(XL, EXPO, DEN, ei_s, OUT, ES, H, HC, shiftC);
    k_bias_relu<<<EW((long)NS * HC), 256, 0, stream>>>(OUT, bias, (long)NS * HC, HC);
    // GraphNorm
    k_fill_f32<<<EW((long)G * HC), 256, 0, stream>>>(GSUM, (long)G * HC, 0.0f);
    k_seg_add<<<EW((long)NS * HC), 256, 0, stream>>>(OUT, bt_s, GSUM, (long)NS * HC, HC);
    k_div_cnt<<<EW((long)G * HC), 256, 0, stream>>>(GSUM, CNT, (long)G * HC, HC);  // -> mean
    k_fill_f32<<<EW((long)G * HC), 256, 0, stream>>>(GVAR, (long)G * HC, 0.0f);
    k_gn_var<<<EW((long)NS * HC), 256, 0, stream>>>(OUT, bt_s, GSUM, gna, GVAR, (long)NS * HC, HC);
    k_gn_apply<<<EW((long)NS * HC), 256, 0, stream>>>(OUT, bt_s, GSUM, gna, gnw, gnb, GVAR, CNT, (long)NS * HC, HC);
  };

  // ---- s-branch ----
  k_convert_pad<<<EW((long)NS * 96), 256, 0, stream>>>(x_s, ABF, NS, FS, 96, (long)NS * 96);
  k_fill_f32<<<EW(G), 256, 0, stream>>>(CNT, G, 0.0f);
  k_counts<<<EW(NS), 256, 0, stream>>>(bt_s, CNT, NS);

  run_gat_layer(69,   96, 8, 128, 7, F(8),  F(9),  F(10), F(11), F(12), F(23), F(24), F(25));
  k_convert_pad<<<EW((long)NS * 1024), 256, 0, stream>>>(XR, ABF, NS, 1024, 1024, (long)NS * 1024);
  run_gat_layer(1024, 1024, 4, 256, 8, F(13), F(14), F(15), F(16), F(17), F(26), F(27), F(28));
  k_convert_pad<<<EW((long)NS * 1024), 256, 0, stream>>>(XR, ABF, NS, 1024, 1024, (long)NS * 1024);
  run_gat_layer(1024, 1024, 1, 512, 9, F(18), F(19), F(20), F(21), F(22), F(29), F(30), F(31));

  // mean pool (Cdim = 512)
  k_fill_f32<<<EW((long)G * 512), 256, 0, stream>>>(POOL, (long)G * 512, 0.0f);
  k_seg_add<<<EW((long)NS * 512), 256, 0, stream>>>(XR, bt_s, POOL, (long)NS * 512, 512);
  k_div_cnt<<<EW((long)G * 512), 256, 0, stream>>>(POOL, CNT, (long)G * 512, 512);

  // ---- head ----
  k_convert_pad<<<EW((long)G * 512), 256, 0, stream>>>(POOL, ABF, G, 512, 512, (long)G * 512);
  k_convert_pad<<<EW((long)512 * 1024), 256, 0, stream>>>(F(56), WBF, 512, 1024, 1024, (long)512 * 1024);
  k_gemm_bf16<<<GG(G, 1024), 256, 0, stream>>>(ABF, WBF, FCH, G, 1024, 1024, 512, 1024);
  k_bias<<<EW((long)G * 1024), 256, 0, stream>>>(FCH, F(57), (long)G * 1024, 1024);
  k_bn_relu_bf16<<<1024, 256, 0, stream>>>(FCH, F(58), F(59), FCHBF, G, 1024);

  k_convert_pad<<<EW((long)1024 * 1408), 256, 0, stream>>>(F(60), WBF, 1024, 1317, 1408, (long)1024 * 1408);
  float* OUTF = (float*)d_out;
  k_gemm_bf16<<<GG(G, 1408), 256, 0, stream>>>(FCHBF, WBF, OUTF, G, 1317, 1408, 1024, 1317);
  k_final<<<EW((long)G * 1317), 256, 0, stream>>>(OUTF, F(61), (long)G * 1317, 1317);
}
